// PIGNN_66580583022911
// MI455X (gfx1250) — compile-verified, matrix operand order checked by reading, not run on hardware
//
#include <hip/hip_runtime.h>
#include <math.h>

// ---------------- CDNA5 WMMA types ----------------
typedef __attribute__((ext_vector_type(16))) __bf16 v16bf;
typedef __attribute__((ext_vector_type(8)))  float  v8f;
typedef __attribute__((ext_vector_type(8)))  unsigned short us8;

union FragU { v16bf bf; us8 h8[2]; };

__device__ __forceinline__ unsigned short f2bf(float f) {
  unsigned int x = __float_as_uint(f);
  unsigned int r = x + 0x7FFFu + ((x >> 16) & 1u);   // round-to-nearest-even
  return (unsigned short)(r >> 16);
}

__device__ __forceinline__ float silu_f(float v) { return v / (1.f + __expf(-v)); }

// load a 16-bf16 WMMA fragment: 8 contiguous halfwords at k, 8 at k+16
__device__ __forceinline__ v16bf ldfrag(const unsigned short* p) {
  const us8* q = (const us8*)__builtin_assume_aligned((const void*)p, 16);
  FragU f; f.h8[0] = q[0]; f.h8[1] = q[2];
  return f.bf;
}

#define LNEPS 1e-5f
#define BNEPS 1e-5f

// ---------------- encoder: h = silu(LN(x@W + b)) ----------------
__global__ __launch_bounds__(64) void enc_kernel(
    const float* __restrict__ x, const float* __restrict__ W,
    const float* __restrict__ bias, const float* __restrict__ g,
    const float* __restrict__ beta, float* __restrict__ h, int N) {
  __shared__ float sW[32 * 128];
  __shared__ float sO[64 * 129];
  __shared__ float sM[64], sR[64];
  const int tid = threadIdx.x;
  const long row0 = (long)blockIdx.x * 64;
  for (int i = tid; i < 32 * 128; i += 64) sW[i] = W[i];
  long n = row0 + tid; long nn = (n < N) ? n : (long)N - 1;
  float xr[32];
#pragma unroll
  for (int k = 0; k < 32; ++k) xr[k] = x[nn * 32 + k];
  __syncthreads();
  float mean = 0.f;
  for (int c = 0; c < 128; ++c) {
    float acc = bias[c];
#pragma unroll
    for (int k = 0; k < 32; ++k) acc += xr[k] * sW[k * 128 + c];
    sO[tid * 129 + c] = acc;
    mean += acc;
  }
  mean *= (1.f / 128.f);
  float var = 0.f;
  for (int c = 0; c < 128; ++c) { float d = sO[tid * 129 + c] - mean; var += d * d; }
  var *= (1.f / 128.f);
  sM[tid] = mean; sR[tid] = rsqrtf(var + LNEPS);
  __syncthreads();
  for (int i = tid; i < 64 * 128; i += 64) {
    int r = i >> 7, c = i & 127;
    long gr = row0 + r;
    if (gr < N) {
      float v = (sO[r * 129 + c] - sM[r]) * sR[r] * g[c] + beta[c];
      h[gr * 128 + c] = silu_f(v);
    }
  }
}

// ---------------- edge scatter: acc[dst] += h[src] (one wave per edge) ----------------
__global__ __launch_bounds__(256) void edge_kernel(
    const int* __restrict__ ei, const float* __restrict__ h,
    float* __restrict__ acc, long E) {
  long gid = (long)blockIdx.x * 256 + threadIdx.x;
  long e = gid >> 5;
  if (e >= E) return;
  int lane = (int)(gid & 31);
  long src = ei[e];
  long dst = ei[E + e];
  float4 v = ((const float4*)(h + src * 128))[lane];
  float* p = acc + dst * 128 + (long)lane * 4;
  atomicAdd(p + 0, v.x); atomicAdd(p + 1, v.y);
  atomicAdd(p + 2, v.z); atomicAdd(p + 3, v.w);
}

// ---------------- fused GIN MLP: z <- silu(z@W1+b1)@W2+b2 (in place), + BN stats ----------------
#define WST 136   // padded LDS row stride (ushorts), multiple of 8 for 16B-aligned frags
__global__ __launch_bounds__(256) void mlp_kernel(
    float* __restrict__ z, const float* __restrict__ W1, const float* __restrict__ b1,
    const float* __restrict__ W2, const float* __restrict__ b2,
    float* __restrict__ stats, int N) {
  __shared__ unsigned short sA[128 * WST];
  __shared__ unsigned short sW[128 * WST];
  const int tid = threadIdx.x;
  const long row0 = (long)blockIdx.x * 128;

  // stage W1^T (bf16) and A tile (bf16, row-major)
  for (int idx = tid; idx < 128 * 128; idx += 256) {
    int k = idx >> 7, c = idx & 127;           // W1[k][c]
    sW[c * WST + k] = f2bf(W1[idx]);
  }
  for (int idx = tid; idx < 128 * 128; idx += 256) {
    int r = idx >> 7, k = idx & 127;
    long row = row0 + r; if (row >= N) row = (long)N - 1;
    sA[r * WST + k] = f2bf(z[row * 128 + k]);
  }
  __syncthreads();

  const int wave = tid >> 5, lane = tid & 31;
  const int lrow = lane & 15, hi = lane >> 4;
  const unsigned short* aBase = &sA[(wave * 16 + lrow) * WST];

  v8f acc[8];
  const v8f vzero = {0.f, 0.f, 0.f, 0.f, 0.f, 0.f, 0.f, 0.f};
#pragma unroll
  for (int j = 0; j < 8; ++j) acc[j] = vzero;

  // GEMM1: 8 col-tiles x 4 k-steps
#pragma unroll
  for (int kk = 0; kk < 4; ++kk) {
    v16bf a = ldfrag(aBase + kk * 32 + hi * 8);
#pragma unroll
    for (int j = 0; j < 8; ++j) {
      v16bf b = ldfrag(&sW[(j * 16 + lrow) * WST + kk * 32 + hi * 8]);
      acc[j] = __builtin_amdgcn_wmma_f32_16x16x32_bf16(
          false, a, false, b, (short)0, acc[j], false, false);
    }
  }
  __syncthreads();

  // m1 = silu(acc + b1) -> sA (bf16); reload sW with W2^T
#pragma unroll
  for (int j = 0; j < 8; ++j) {
    int col = j * 16 + lrow;
    float bb = b1[col];
#pragma unroll
    for (int r = 0; r < 8; ++r) {
      float v = acc[j][r] + bb;
      sA[(wave * 16 + r + 8 * hi) * WST + col] = f2bf(silu_f(v));
    }
  }
  for (int idx = tid; idx < 128 * 128; idx += 256) {
    int k = idx >> 7, c = idx & 127;
    sW[c * WST + k] = f2bf(W2[idx]);
  }
  __syncthreads();

  // GEMM2
#pragma unroll
  for (int j = 0; j < 8; ++j) acc[j] = vzero;
#pragma unroll
  for (int kk = 0; kk < 4; ++kk) {
    v16bf a = ldfrag(aBase + kk * 32 + hi * 8);
#pragma unroll
    for (int j = 0; j < 8; ++j) {
      v16bf b = ldfrag(&sW[(j * 16 + lrow) * WST + kk * 32 + hi * 8]);
      acc[j] = __builtin_amdgcn_wmma_f32_16x16x32_bf16(
          false, a, false, b, (short)0, acc[j], false, false);
    }
  }
  __syncthreads();

  // epilogue: z2 = acc + b2 written in place; per-channel sum/sumsq for BN
  float* sStat = (float*)sA;   // sA no longer needed
  if (tid < 256) sStat[tid] = 0.f;
  __syncthreads();
#pragma unroll
  for (int j = 0; j < 8; ++j) {
    int col = j * 16 + lrow;
    float bb = b2[col];
    float s1 = 0.f, s2 = 0.f;
#pragma unroll
    for (int r = 0; r < 8; ++r) {
      long grow = row0 + wave * 16 + r + 8 * hi;
      float v = acc[j][r] + bb;
      if (grow < N) {
        z[grow * 128 + col] = v;
        s1 += v; s2 += v * v;
      }
    }
    atomicAdd(&sStat[col], s1);
    atomicAdd(&sStat[128 + col], s2);
  }
  __syncthreads();
  if (tid < 256) atomicAdd(&stats[tid], sStat[tid]);
}

// ---------------- BN finalize: h = silu(bn(z2)) ----------------
__global__ __launch_bounds__(256) void bn_kernel(
    const float* __restrict__ z2, float* __restrict__ h,
    const float* __restrict__ stats, const float* __restrict__ g,
    const float* __restrict__ beta, long total, float invN) {
  long i = (long)blockIdx.x * 256 + threadIdx.x;
  if (i >= total) return;
  int c = (int)(i & 127);
  float m = stats[c] * invN;
  float v = stats[128 + c] * invN - m * m;
  float val = (z2[i] - m) * rsqrtf(v + BNEPS) * g[c] + beta[c];
  h[i] = silu_f(val);
}

// ---------------- global mean pool (sums + counts) ----------------
__global__ __launch_bounds__(256) void pool_kernel(
    const float* __restrict__ h, const int* __restrict__ batch,
    float* __restrict__ pool, float* __restrict__ cnt, long N) {
  long gid = (long)blockIdx.x * 256 + threadIdx.x;
  long n = gid >> 5;
  if (n >= N) return;
  int lane = (int)(gid & 31);
  long b = batch[n];
  float4 v = ((const float4*)(h + n * 128))[lane];
  float* p = pool + b * 128 + (long)lane * 4;
  atomicAdd(p + 0, v.x); atomicAdd(p + 1, v.y);
  atomicAdd(p + 2, v.z); atomicAdd(p + 3, v.w);
  if (lane == 0) atomicAdd(&cnt[b], 1.f);
}

// ---------------- fusion head: one block per graph ----------------
__device__ __forceinline__ float2 blockReduce2(float a, float b, float* s1, float* s2) {
  int tid = threadIdx.x;
  s1[tid] = a; s2[tid] = b;
  __syncthreads();
  for (int off = 128; off > 0; off >>= 1) {
    if (tid < off) { s1[tid] += s1[tid + off]; s2[tid] += s2[tid + off]; }
    __syncthreads();
  }
  float2 r; r.x = s1[0]; r.y = s2[0];
  __syncthreads();
  return r;
}

__global__ __launch_bounds__(256) void fusion_kernel(
    const float* __restrict__ pool, const float* __restrict__ cnt,
    const float* __restrict__ u,
    const float* __restrict__ Wp, const float* __restrict__ bp,
    const float* __restrict__ gp, const float* __restrict__ betap,
    const float* __restrict__ W1, const float* __restrict__ b1,
    const float* __restrict__ g1, const float* __restrict__ beta1,
    const float* __restrict__ W2, const float* __restrict__ b2,
    const float* __restrict__ g2, const float* __restrict__ beta2,
    const float* __restrict__ Wt, const float* __restrict__ bt,
    const float* __restrict__ Wz, const float* __restrict__ bz,
    float* __restrict__ out_theta, float* __restrict__ out_z) {
  __shared__ float f[256], t2[256], s1[256], s2[256];
  const int b = blockIdx.x, tid = threadIdx.x;

  // h_graph into f[0:128]
  if (tid < 128) {
    float c = fmaxf(cnt[b], 1.f);
    f[tid] = pool[(long)b * 128 + tid] / c;
  }
  // phys raw
  float praw = 0.f;
  if (tid >= 128) {
    int c = tid - 128;
    float a = bp[c];
#pragma unroll
    for (int k = 0; k < 16; ++k) a += u[b * 16 + k] * Wp[k * 128 + c];
    praw = a;
  }
  float2 mv = blockReduce2(tid >= 128 ? praw : 0.f, tid >= 128 ? praw * praw : 0.f, s1, s2);
  if (tid >= 128) {
    float m = mv.x / 128.f, v = mv.y / 128.f - m * m;
    float x = (praw - m) * rsqrtf(v + LNEPS) * gp[tid - 128] + betap[tid - 128];
    f[tid] = silu_f(x);
  }
  __syncthreads();

  const int c = tid;
  // fus1
  float a1 = b1[c];
  for (int k = 0; k < 256; ++k) a1 += f[k] * W1[k * 256 + c];
  mv = blockReduce2(a1, a1 * a1, s1, s2);
  {
    float m = mv.x / 256.f, v = mv.y / 256.f - m * m;
    float x = (a1 - m) * rsqrtf(v + LNEPS) * g1[c] + beta1[c];
    t2[c] = silu_f(x);
  }
  __syncthreads();
  // fus2
  float a2 = b2[c];
  for (int k = 0; k < 256; ++k) a2 += t2[k] * W2[k * 256 + c];
  mv = blockReduce2(a2, a2 * a2, s1, s2);
  {
    float m = mv.x / 256.f, v = mv.y / 256.f - m * m;
    float x = (a2 - m) * rsqrtf(v + LNEPS) * g2[c] + beta2[c];
    f[c] = silu_f(x);
  }
  __syncthreads();
  // heads
  if (c < 10) {
    float a = bt[c];
    for (int k = 0; k < 256; ++k) a += f[k] * Wt[k * 10 + c];
    out_theta[(long)b * 10 + c] = a;
  }
  if (c < 128) {
    float a = bz[c];
    for (int k = 0; k < 256; ++k) a += f[k] * Wz[k * 128 + c];
    out_z[(long)b * 128 + c] = a;
  }
}

// ---------------- launch ----------------
extern "C" void kernel_launch(void* const* d_in, const int* in_sizes, int n_in,
                              void* d_out, int out_size, void* d_ws, size_t ws_size,
                              hipStream_t stream) {
  const float* x    = (const float*)d_in[0];
  const int*  edge  = (const int*)d_in[1];     // (2,E): [0:E)=src, [E:2E)=dst
  const int*  batch = (const int*)d_in[2];
  const float* u    = (const float*)d_in[3];
  const long N = in_sizes[0] / 32;
  const long E = in_sizes[1] / 2;
  const int  B = in_sizes[3] / 16;

  // params flattened in jax tree order (sorted dict keys):
  // bns[l]: b,g ; convs[l]: b1,b2,w1,w2 ; enc: b,ln_b,ln_g,w ; fus1/fus2: b,ln_b,ln_g,w ;
  // phys: b,ln_b,ln_g,w ; theta: b,w ; z: b,w
  const float* bn_b[5]; const float* bn_g[5];
  for (int l = 0; l < 5; ++l) { bn_b[l] = (const float*)d_in[4 + 2 * l];
                                bn_g[l] = (const float*)d_in[5 + 2 * l]; }
  const float* cv_b1[5]; const float* cv_b2[5]; const float* cv_w1[5]; const float* cv_w2[5];
  for (int l = 0; l < 5; ++l) {
    cv_b1[l] = (const float*)d_in[14 + 4 * l];
    cv_b2[l] = (const float*)d_in[15 + 4 * l];
    cv_w1[l] = (const float*)d_in[16 + 4 * l];
    cv_w2[l] = (const float*)d_in[17 + 4 * l];
  }
  const float* enc_b   = (const float*)d_in[34];
  const float* enc_lnb = (const float*)d_in[35];
  const float* enc_lng = (const float*)d_in[36];
  const float* enc_w   = (const float*)d_in[37];
  const float* f1_b    = (const float*)d_in[38];
  const float* f1_lnb  = (const float*)d_in[39];
  const float* f1_lng  = (const float*)d_in[40];
  const float* f1_w    = (const float*)d_in[41];
  const float* f2_b    = (const float*)d_in[42];
  const float* f2_lnb  = (const float*)d_in[43];
  const float* f2_lng  = (const float*)d_in[44];
  const float* f2_w    = (const float*)d_in[45];
  const float* ph_b    = (const float*)d_in[46];
  const float* ph_lnb  = (const float*)d_in[47];
  const float* ph_lng  = (const float*)d_in[48];
  const float* ph_w    = (const float*)d_in[49];
  const float* th_b    = (const float*)d_in[50];
  const float* th_w    = (const float*)d_in[51];
  const float* zz_b    = (const float*)d_in[52];
  const float* zz_w    = (const float*)d_in[53];

  // workspace carve-up
  float* h     = (float*)d_ws;                 // N*128
  float* tmp   = h + N * 128;                  // N*128
  float* stats = tmp + N * 128;                // 256
  float* pool  = stats + 256;                  // B*128
  float* cnt   = pool + (long)B * 128;         // B

  float* out_theta = (float*)d_out;
  float* out_z     = out_theta + (long)B * 10;

  hipMemsetAsync(pool, 0, ((size_t)B * 128 + B) * sizeof(float), stream);

  // encoder
  {
    int blocks = (int)((N + 63) / 64);
    enc_kernel<<<blocks, 64, 0, stream>>>(x, enc_w, enc_b, enc_lng, enc_lnb, h, (int)N);
  }

  // GIN layers
  for (int l = 0; l < 5; ++l) {
    hipMemcpyAsync(tmp, h, (size_t)N * 128 * sizeof(float),
                   hipMemcpyDeviceToDevice, stream);
    {
      int blocks = (int)((E * 32 + 255) / 256);
      edge_kernel<<<blocks, 256, 0, stream>>>(edge, h, tmp, E);
    }
    hipMemsetAsync(stats, 0, 256 * sizeof(float), stream);
    {
      int blocks = (int)((N + 127) / 128);
      mlp_kernel<<<blocks, 256, 0, stream>>>(tmp, cv_w1[l], cv_b1[l], cv_w2[l],
                                             cv_b2[l], stats, (int)N);
    }
    {
      long total = N * 128;
      int blocks = (int)((total + 255) / 256);
      bn_kernel<<<blocks, 256, 0, stream>>>(tmp, h, stats, bn_g[l], bn_b[l],
                                            total, 1.f / (float)N);
    }
  }

  // pooling + fusion head
  {
    int blocks = (int)((N * 32 + 255) / 256);
    pool_kernel<<<blocks, 256, 0, stream>>>(h, batch, pool, cnt, N);
  }
  fusion_kernel<<<B, 256, 0, stream>>>(pool, cnt, u,
                                       ph_w, ph_b, ph_lng, ph_lnb,
                                       f1_w, f1_b, f1_lng, f1_lnb,
                                       f2_w, f2_b, f2_lng, f2_lnb,
                                       th_w, th_b, zz_w, zz_b,
                                       out_theta, out_z);
}
